// GPT2SelfAttention_19490561589680
// MI455X (gfx1250) — matrix operand order repaired and checked by reading, not compiled, not run on hardware
//
#include <hip/hip_runtime.h>
#include <hip/hip_bf16.h>

typedef __attribute__((ext_vector_type(8)))  _Float16 v8h;
typedef __attribute__((ext_vector_type(16))) _Float16 v16h;
typedef __attribute__((ext_vector_type(8)))  float    v8f;

#define B_    2
#define S_    2048
#define D_    1024
#define H_    16
#define HD_   64
#define M_TOT (B_ * S_)   // 4096 rows
#define N_QKV (3 * D_)    // 3072

union AF { v16h v; v8h h[2]; };

__device__ __forceinline__ v16h load_frag2(const _Float16* p0, const _Float16* p1) {
    AF f;
    f.h[0] = *(const v8h*)p0;
    f.h[1] = *(const v8h*)p1;
    return f.v;
}

__device__ __forceinline__ v8f wmma_f16(v16h a, v16h b, v8f c) {
    return __builtin_amdgcn_wmma_f32_16x16x32_f16(false, a, false, b, (short)0, c,
                                                  false, false);
}

// ---------------------------------------------------------------------------
// Wave-level GEMM core: one wave computes a 16x64 tile of A[M,K] * Bt[N,K]^T.
// A row-major (lda = K), Bt stores B transposed row-major by N (ldb = K) so
// every fragment is two contiguous b128 loads per lane.
// ---------------------------------------------------------------------------
__device__ __forceinline__ void gemm16x64(const _Float16* __restrict__ A, int lda,
                                          const _Float16* __restrict__ Bt, int ldb,
                                          int m0, int n0, int K, int lane,
                                          v8f acc[4]) {
    const int col = lane & 15;
    const int hi  = lane >> 4;
    // A 16x32 f16 fragment: lanes 0-15 hold row=lane, K kk..kk+15;
    // lanes 16-31 hold row=lane-16, K kk+16..kk+31.
    const _Float16* arow = A + (size_t)(m0 + col) * lda + hi * 16;
    const _Float16* brow0 = Bt + (size_t)(n0 +  0 + col) * ldb + hi * 16;
    const _Float16* brow1 = Bt + (size_t)(n0 + 16 + col) * ldb + hi * 16;
    const _Float16* brow2 = Bt + (size_t)(n0 + 32 + col) * ldb + hi * 16;
    const _Float16* brow3 = Bt + (size_t)(n0 + 48 + col) * ldb + hi * 16;
    for (int kk = 0; kk < K; kk += 32) {
        __builtin_prefetch(arow + kk + 128, 0, 0);
        v16h a = load_frag2(arow + kk, arow + kk + 8);
        // B 32x16 fragment: lanes 0-15 N=lane, K 0..15; lanes 16-31 N=lane-16, K 16..31
        v16h b0 = load_frag2(brow0 + kk, brow0 + kk + 8);
        acc[0] = wmma_f16(a, b0, acc[0]);
        v16h b1 = load_frag2(brow1 + kk, brow1 + kk + 8);
        acc[1] = wmma_f16(a, b1, acc[1]);
        v16h b2 = load_frag2(brow2 + kk, brow2 + kk + 8);
        acc[2] = wmma_f16(a, b2, acc[2]);
        v16h b3 = load_frag2(brow3 + kk, brow3 + kk + 8);
        acc[3] = wmma_f16(a, b3, acc[3]);
    }
}

// ---------------------------------------------------------------------------
// Precision conversion kernels
// ---------------------------------------------------------------------------
__global__ void cvt_f32_f16(const float* __restrict__ src, _Float16* __restrict__ dst,
                            int n) {
    for (int i = blockIdx.x * blockDim.x + threadIdx.x; i < n;
         i += gridDim.x * blockDim.x)
        dst[i] = (_Float16)src[i];
}

// Wt[n * rows + k] = (f16) W[k * cols + n]   (rows = K dim, cols = N dim)
__global__ void transpose_cvt(const float* __restrict__ W, _Float16* __restrict__ Wt,
                              int rows, int cols) {
    int total = rows * cols;
    for (int i = blockIdx.x * blockDim.x + threadIdx.x; i < total;
         i += gridDim.x * blockDim.x) {
        int n = i / rows;
        int k = i - n * rows;
        Wt[i] = (_Float16)W[(size_t)k * cols + n];
    }
}

// ---------------------------------------------------------------------------
// Fused QKV projection: qkv = xh @ Wqkv + b, scattered to head-major layouts.
// Q scaled by 1/sqrt(hd). Q,K: [B,H,S,hd] f16.  V: transposed [B,H,hd,S] f16.
// ---------------------------------------------------------------------------
__global__ void qkv_gemm_kernel(const _Float16* __restrict__ xh,
                                const _Float16* __restrict__ WqkvT,
                                const float* __restrict__ bqkv,
                                _Float16* __restrict__ Qb,
                                _Float16* __restrict__ Kb,
                                _Float16* __restrict__ Vt) {
    const int lane = threadIdx.x & 31;
    const int wave = blockIdx.x * (blockDim.x >> 5) + (threadIdx.x >> 5);
    const int nNG = N_QKV / 64;                 // 48
    const int nTask = (M_TOT / 16) * nNG;       // 12288
    if (wave >= nTask) return;
    const int mt = wave / nNG;
    const int ng = wave - mt * nNG;
    const int m0 = mt * 16, n0 = ng * 64;

    v8f acc[4];
    const v8f zero8 = {0.f, 0.f, 0.f, 0.f, 0.f, 0.f, 0.f, 0.f};
#pragma unroll
    for (int t = 0; t < 4; ++t) acc[t] = zero8;

    gemm16x64(xh, D_, WqkvT, D_, m0, n0, D_, lane, acc);

    const int col = lane & 15, hi = lane >> 4;
#pragma unroll
    for (int t = 0; t < 4; ++t) {
        const int n = n0 + t * 16 + col;
        const float bias = bqkv[n];
        const int which = n / D_;               // 0=Q 1=K 2=V (uniform per wave)
        const int nd = n - which * D_;
        const int h  = nd >> 6;
        const int dd = nd & (HD_ - 1);
#pragma unroll
        for (int r = 0; r < 8; ++r) {
            const int m = m0 + r + hi * 8;      // row in [B*S)
            const int b = m >> 11;
            const int s = m & (S_ - 1);
            const size_t hb = (size_t)(b * H_ + h);
            const float v = acc[t][r] + bias;
            if (which == 0)
                Qb[(hb * S_ + s) * HD_ + dd] = (_Float16)(v * 0.125f); // 1/sqrt(64)
            else if (which == 1)
                Kb[(hb * S_ + s) * HD_ + dd] = (_Float16)v;
            else
                Vt[(hb * HD_ + dd) * S_ + s] = (_Float16)v;
        }
    }
}

// ---------------------------------------------------------------------------
// Flash-style causal attention: one wave per (b, h, 16-row q-tile).
// Key blocks of 32: 4 WMMA for scores + 4 WMMA for P@V per block.
// Writes merged-head output Ot [B*S, D] f16.
// ---------------------------------------------------------------------------
__global__ void attn_kernel(const _Float16* __restrict__ Qb,
                            const _Float16* __restrict__ Kb,
                            const _Float16* __restrict__ Vt,
                            _Float16* __restrict__ Ot) {
    __shared__ _Float16 lds[8][16][32] __attribute__((aligned(128)));

    const int lane = threadIdx.x & 31;
    const int wid  = threadIdx.x >> 5;
    const int wave = blockIdx.x * (blockDim.x >> 5) + wid;
    const int nQT = S_ / 16;                    // 128
    const int nTask = B_ * H_ * nQT;            // 4096
    if (wave >= nTask) return;

    const int bh = wave / nQT;
    const int qi = wave - bh * nQT;
    const int q0 = qi * 16;

    const _Float16* Qh = Qb + (size_t)bh * S_ * HD_;
    const _Float16* Kh = Kb + (size_t)bh * S_ * HD_;
    const _Float16* Vh = Vt + (size_t)bh * HD_ * S_;

    const int col = lane & 15, hi = lane >> 4;

    // Q fragments for this q-tile (hd=64 -> two chained k=32 fragments)
    const _Float16* qrow = Qh + (size_t)(q0 + col) * HD_ + hi * 16;
    const v16h qa0 = load_frag2(qrow + 0,  qrow + 8);
    const v16h qa1 = load_frag2(qrow + 32, qrow + 40);

    const v8f zero8 = {0.f, 0.f, 0.f, 0.f, 0.f, 0.f, 0.f, 0.f};
    v8f o[4];
#pragma unroll
    for (int t = 0; t < 4; ++t) o[t] = zero8;
    float mr[8], lr[8];
#pragma unroll
    for (int r = 0; r < 8; ++r) { mr[r] = -3.0e38f; lr[r] = 0.0f; }

    const int nKB = (q0 + 15) / 32 + 1;         // key blocks covering causal span
    for (int j = 0; j < nKB; ++j) {
        const int k0 = j * 32;

        // ---- scores S = Q @ K^T (already scaled via Q) ----
        v8f sc[2];
#pragma unroll
        for (int kt = 0; kt < 2; ++kt) {
            const _Float16* krow = Kh + (size_t)(k0 + kt * 16 + col) * HD_ + hi * 16;
            v16h kb0 = load_frag2(krow + 0,  krow + 8);
            v16h kb1 = load_frag2(krow + 32, krow + 40);
            v8f s = zero8;
            s = wmma_f16(qa0, kb0, s);
            s = wmma_f16(qa1, kb1, s);
            sc[kt] = s;
        }

        // ---- causal mask (reference: fill with -10000 before softmax) ----
        if (k0 + 31 > q0) {
#pragma unroll
            for (int kt = 0; kt < 2; ++kt)
#pragma unroll
                for (int r = 0; r < 8; ++r) {
                    const int row = q0 + r + hi * 8;
                    const int c   = k0 + kt * 16 + col;
                    if (c > row) sc[kt][r] = -10000.0f;
                }
        }

        // ---- online softmax; row lives on 16 lanes of a half-wave ----
#pragma unroll
        for (int r = 0; r < 8; ++r) {
            float bm = fmaxf(sc[0][r], sc[1][r]);
#pragma unroll
            for (int off = 8; off >= 1; off >>= 1)
                bm = fmaxf(bm, __shfl_xor(bm, off, 16));
            const float mnew = fmaxf(mr[r], bm);
            const float p0 = __expf(sc[0][r] - mnew);
            const float p1 = __expf(sc[1][r] - mnew);
            float rs = p0 + p1;
#pragma unroll
            for (int off = 8; off >= 1; off >>= 1)
                rs += __shfl_xor(rs, off, 16);
            const float co = __expf(mr[r] - mnew);
            lr[r] = lr[r] * co + rs;
            mr[r] = mnew;
#pragma unroll
            for (int t = 0; t < 4; ++t) o[t][r] *= co;
            // stage P (C-layout -> row-major 16x32) for the A-fragment reload
            lds[wid][r + hi * 8][col]      = (_Float16)p0;
            lds[wid][r + hi * 8][16 + col] = (_Float16)p1;
        }

        __builtin_amdgcn_wave_barrier();
        asm volatile("s_wait_dscnt 0" ::: "memory");
        __builtin_amdgcn_wave_barrier();

        // reload P as 16x32 A-fragment
        const _Float16* prow = &lds[wid][col][hi * 16];
        const v16h pa = load_frag2(prow, prow + 8);

        // ---- O += P @ V ----
#pragma unroll
        for (int t = 0; t < 4; ++t) {
            const _Float16* vrow = Vh + (size_t)(t * 16 + col) * S_ + k0 + hi * 16;
            v16h vb = load_frag2(vrow, vrow + 8);
            o[t] = wmma_f16(pa, vb, o[t]);
        }
    }

    // ---- normalize and write merged-head output ----
    const int b = bh / H_;
    const int h = bh - b * H_;
#pragma unroll
    for (int r = 0; r < 8; ++r) {
        const float inv = 1.0f / lr[r];
        const size_t m = (size_t)b * S_ + q0 + r + hi * 8;
#pragma unroll
        for (int t = 0; t < 4; ++t)
            Ot[m * D_ + h * HD_ + t * 16 + col] = (_Float16)(o[t][r] * inv);
    }
}

// ---------------------------------------------------------------------------
// Output projection: out = Ot @ W_out + b_out  (f32 result)
// ---------------------------------------------------------------------------
__global__ void out_gemm_kernel(const _Float16* __restrict__ Ot,
                                const _Float16* __restrict__ WoutT,
                                const float* __restrict__ bout,
                                float* __restrict__ out) {
    const int lane = threadIdx.x & 31;
    const int wave = blockIdx.x * (blockDim.x >> 5) + (threadIdx.x >> 5);
    const int nNG = D_ / 64;                    // 16
    const int nTask = (M_TOT / 16) * nNG;       // 4096
    if (wave >= nTask) return;
    const int mt = wave / nNG;
    const int ng = wave - mt * nNG;
    const int m0 = mt * 16, n0 = ng * 64;

    v8f acc[4];
    const v8f zero8 = {0.f, 0.f, 0.f, 0.f, 0.f, 0.f, 0.f, 0.f};
#pragma unroll
    for (int t = 0; t < 4; ++t) acc[t] = zero8;

    gemm16x64(Ot, D_, WoutT, D_, m0, n0, D_, lane, acc);

    const int col = lane & 15, hi = lane >> 4;
#pragma unroll
    for (int t = 0; t < 4; ++t) {
        const int n = n0 + t * 16 + col;
        const float bias = bout[n];
#pragma unroll
        for (int r = 0; r < 8; ++r) {
            const size_t m = (size_t)(m0 + r + hi * 8);
            out[m * D_ + n] = acc[t][r] + bias;
        }
    }
}

// ---------------------------------------------------------------------------
// Launch
// ---------------------------------------------------------------------------
extern "C" void kernel_launch(void* const* d_in, const int* in_sizes, int n_in,
                              void* d_out, int out_size, void* d_ws, size_t ws_size,
                              hipStream_t stream) {
    (void)in_sizes; (void)n_in; (void)out_size; (void)ws_size;
    const float* x    = (const float*)d_in[0];
    const float* Wqkv = (const float*)d_in[1];
    const float* bqkv = (const float*)d_in[2];
    const float* Wout = (const float*)d_in[3];
    const float* bout = (const float*)d_in[4];
    float* out = (float*)d_out;

    char* ws = (char*)d_ws;
    const size_t MB = 1u << 20;
    _Float16* xh    = (_Float16*)(ws +  0 * MB);  // 8 MB  [B*S, D] f16
    _Float16* WqkvT = (_Float16*)(ws +  8 * MB);  // 6 MB  [3D, D] f16 (transposed)
    _Float16* WoutT = (_Float16*)(ws + 14 * MB);  // 2 MB  [D, D]  f16 (transposed)
    _Float16* Qb    = (_Float16*)(ws + 16 * MB);  // 8 MB  [B,H,S,hd] (pre-scaled)
    _Float16* Kb    = (_Float16*)(ws + 24 * MB);  // 8 MB  [B,H,S,hd]
    _Float16* Vt    = (_Float16*)(ws + 32 * MB);  // 8 MB  [B,H,hd,S]
    _Float16* Ot    = (_Float16*)(ws + 40 * MB);  // 8 MB  [B*S, D]

    cvt_f32_f16<<<4096, 256, 0, stream>>>(x, xh, M_TOT * D_);
    transpose_cvt<<<4096, 256, 0, stream>>>(Wqkv, WqkvT, D_, N_QKV);
    transpose_cvt<<<2048, 256, 0, stream>>>(Wout, WoutT, D_, D_);

    // 12288 wave-tiles, 8 waves (wave32) per 256-thread block
    qkv_gemm_kernel<<<1536, 256, 0, stream>>>(xh, WqkvT, bqkv, Qb, Kb, Vt);
    // 4096 (b, h, q-tile) tasks
    attn_kernel<<<512, 256, 0, stream>>>(Qb, Kb, Vt, Ot);
    // 4096 wave-tiles
    out_gemm_kernel<<<512, 256, 0, stream>>>(Ot, WoutT, bout, out);
}
